// GRUEncoder_12446815224031
// MI455X (gfx1250) — matrix operand order repaired and checked by reading, not compiled, not run on hardware
//
#include <hip/hip_runtime.h>

// ---------------------------------------------------------------------------
// GRU encoder for MI455X (gfx1250, wave32, WMMA 16x16x32 f16, async->LDS).
// B=512 T=512 D_X=64 D_U=16 H=128 (3H=384) S=E=64.
// ---------------------------------------------------------------------------

typedef __attribute__((ext_vector_type(16))) _Float16 v16h;
typedef __attribute__((ext_vector_type(8)))  _Float16 v8h;
typedef __attribute__((ext_vector_type(8)))  float    v8f;

constexpr int B_  = 512;
constexpr int T_  = 512;
constexpr int H_  = 128;
constexpr int H3  = 384;
constexpr long M_ = (long)B_ * T_;   // 262144 rows

// ---- WMMA fragment helpers (CDNA5 wave32 layouts, CDNA5 ISA) ----
__device__ __forceinline__ v16h load_a16(const _Float16* base, int ldk,
                                         int lane, int kbase) {
  int row  = lane & 15;
  int koff = (lane < 16) ? 0 : 16;
  const _Float16* p = base + row * ldk + kbase + koff;
  v8h lo = *(const v8h*)p;
  v8h hi = *(const v8h*)(p + 8);
  v16h a;
#pragma unroll
  for (int i = 0; i < 8; ++i) { a[i] = lo[i]; a[8 + i] = hi[i]; }
  return a;
}

__device__ __forceinline__ v16h load_b16(const _Float16* W, int ldk,
                                         int nbase, int kbase, int lane) {
  int n    = lane & 15;
  int koff = (lane < 16) ? 0 : 16;
  return *(const v16h*)(W + (nbase + n) * ldk + kbase + koff);
}

#define WMMA_F16(a, b, c) \
  __builtin_amdgcn_wmma_f32_16x16x32_f16(false, (a), false, (b), (short)0, (c), false, false)

// ---- async global->LDS (CDNA5 ISA, ASYNCcnt) --------
// One instruction copies 16B per lane: LDS[vdst[lane]] = MEM[vaddr[lane]].
__device__ __forceinline__ void async_b128_to_lds(unsigned lds_byte_off,
                                                  const void* gaddr) {
  asm volatile("global_load_async_to_lds_b128 %0, %1, off"
               :: "v"(lds_byte_off), "v"(gaddr)
               : "memory");
}
__device__ __forceinline__ void wait_async0() {
  asm volatile("s_wait_asynccnt 0x0" ::: "memory");
}
__device__ __forceinline__ void wait_async6() {
  asm volatile("s_wait_asynccnt 0x6" ::: "memory");
}
// LDS byte offset of a shared-memory pointer: flat LDS addresses keep the
// LDS offset in the low 32 bits (aperture truncation, CDNA5 ISA).
__device__ __forceinline__ unsigned lds_off(const void* p) {
  return (unsigned)(unsigned long long)p;
}

// ---------------------------------------------------------------------------
// Weight conversion: f32 -> f16, optional K padding (W_in: 80 -> 96).
// ---------------------------------------------------------------------------
__global__ void k_cvt(const float* __restrict__ s, _Float16* __restrict__ d, int n) {
  int i = blockIdx.x * blockDim.x + threadIdx.x;
  if (i < n) d[i] = (_Float16)s[i];
}

__global__ void k_cvt_pad(const float* __restrict__ s, _Float16* __restrict__ d,
                          int rows, int ik, int ok) {
  int i = blockIdx.x * blockDim.x + threadIdx.x;
  int n = rows * ok;
  if (i < n) {
    int r = i / ok, c = i - r * ok;
    d[i] = (c < ik) ? (_Float16)s[r * ik + c] : (_Float16)0.0f;
  }
}

// ---------------------------------------------------------------------------
// xu = concat(x,u) @ W_in.T + b_in  -> f16 [M,128].  K padded 80->96.
// ---------------------------------------------------------------------------
__global__ void k_xu(const float* __restrict__ x, const float* __restrict__ u,
                     const _Float16* __restrict__ Win16, const float* __restrict__ bin,
                     _Float16* __restrict__ xu) {
  __shared__ _Float16 at[16 * 96];
  int tid = threadIdx.x, lane = tid & 31, w = tid >> 5;
  long m0 = (long)blockIdx.x * 16;

  for (int i = tid; i < 16 * 96; i += 128) {
    int r = i / 96, c = i - r * 96;
    long m = m0 + r;
    float v = 0.0f;
    if (c < 64)      v = x[m * 64 + c];
    else if (c < 80) v = u[m * 16 + (c - 64)];
    at[i] = (_Float16)v;
  }
  __syncthreads();

  v8f acc[2] = {};
#pragma unroll
  for (int k = 0; k < 3; ++k) {
    v16h a = load_a16(at, 96, lane, k * 32);
#pragma unroll
    for (int i = 0; i < 2; ++i) {
      int nb = (w + i * 4) * 16;
      v16h b = load_b16(Win16, 96, nb, k * 32, lane);
      acc[i] = WMMA_F16(a, b, acc[i]);
    }
  }
  int n = lane & 15, mo = (lane < 16) ? 0 : 8;
#pragma unroll
  for (int i = 0; i < 2; ++i) {
    int nb = (w + i * 4) * 16;
    float bias = bin[nb + n];
#pragma unroll
    for (int r = 0; r < 8; ++r)
      xu[(m0 + mo + r) * 128 + nb + n] = (_Float16)(acc[i][r] + bias);
  }
}

// ---------------------------------------------------------------------------
// gx = hin(f16 [M,128]) @ W.T + bias -> f32 [M,384]. Throughput GEMM.
// ---------------------------------------------------------------------------
__global__ void k_gx(const _Float16* __restrict__ hin, const _Float16* __restrict__ W16,
                     const float* __restrict__ bias, float* __restrict__ gx) {
  int tid = threadIdx.x, lane = tid & 31, w = tid >> 5;
  long m0  = (long)blockIdx.x * 64 + (long)(w & 3) * 16;
  int  ntb = (w >> 2) * 12;

  const _Float16* arow = hin + m0 * 128;
  v8f acc[12] = {};
#pragma unroll
  for (int k = 0; k < 4; ++k) {
    v16h a = load_a16(arow, 128, lane, k * 32);
#pragma unroll
    for (int i = 0; i < 12; ++i) {
      v16h b = load_b16(W16, 128, (ntb + i) * 16, k * 32, lane);
      acc[i] = WMMA_F16(a, b, acc[i]);
    }
  }
  int n = lane & 15, mo = (lane < 16) ? 0 : 8;
#pragma unroll
  for (int i = 0; i < 12; ++i) {
    int nb = (ntb + i) * 16;
    float bv = bias[nb + n];
#pragma unroll
    for (int r = 0; r < 8; ++r)
      gx[(m0 + mo + r) * H3 + nb + n] = acc[i][r] + bv;
  }
}

// ---------------------------------------------------------------------------
// Recurrent scan over T. One block = 16 batch rows (32 blocks), 256 threads.
// Dynamic LDS (180 KB): Whh f16 (96K) + gh f32 (24K) + h32 (8K) + h16 (4K)
// + double-buffered gx staging (2 x 24K) filled by async global->LDS B128.
// Per step: gh = h @ Whh.T fully from LDS (12 WMMA/wave), async prefetch of
// next step's gx overlaps the WMMA; s_wait_asynccnt retires in order.
// ---------------------------------------------------------------------------
__global__ void k_gru(const float* __restrict__ gx, const _Float16* __restrict__ Whh16g,
                      const float* __restrict__ bhh, _Float16* __restrict__ hseq16,
                      float* __restrict__ hseq32) {
  extern __shared__ char smem[];
  _Float16* Wl  = (_Float16*)smem;                            // 98304 B
  float*    gh  = (float*)(smem + 98304);                     // 24576 B
  float*    h32 = (float*)(smem + 98304 + 24576);             //  8192 B
  _Float16* h16 = (_Float16*)(smem + 98304 + 24576 + 8192);   //  4096 B
  float*    gxs = (float*)(smem + 98304 + 24576 + 8192 + 4096); // 2*24576 B

  int tid = threadIdx.x, lane = tid & 31, w = tid >> 5;
  int b0  = blockIdx.x * 16;

  // ---- stage Whh (f16, 96KB) into LDS via async copies: 6144 x 16B ----
  {
    unsigned wbase = lds_off(Wl);
#pragma unroll
    for (int q = 0; q < 24; ++q) {
      int c = tid + q * 256;                       // 16B chunk index
      async_b128_to_lds(wbase + (unsigned)c * 16, (const void*)(Whh16g + c * 8));
    }
  }
  for (int i = tid; i < 16 * H_; i += 256) { h32[i] = 0.0f; h16[i] = (_Float16)0.0f; }

  // ---- pre-stage gx for t=0 into buffer 0: 1536 x 16B per block ----
  unsigned gxs_base = lds_off(gxs);
  {
#pragma unroll
    for (int q = 0; q < 6; ++q) {
      int c = tid + q * 256;                       // 16B chunk, row-major
      int row = c / 96, col = c - row * 96;
      const float* g = gx + ((long)(b0 + row) * T_ + 0) * H3 + (long)col * 4;
      async_b128_to_lds(gxs_base + (unsigned)c * 16, (const void*)g);
    }
  }
  wait_async0();
  __syncthreads();   // Whh + h init + gx(t=0) visible to all waves

  int erow = tid >> 4;          // 0..15
  int ecol = (tid & 15) * 8;    // 8 cols per thread
  long hrow = (long)(b0 + erow) * T_ * H_;

  for (int t = 0; t < T_; ++t) {
    // ---- issue async staging of gx for t+1 (overlaps WMMA below) ----
    if (t + 1 < T_) {
      unsigned dbase = gxs_base + (unsigned)(((t + 1) & 1) * 24576);
#pragma unroll
      for (int q = 0; q < 6; ++q) {
        int c = tid + q * 256;
        int row = c / 96, col = c - row * 96;
        const float* g = gx + ((long)(b0 + row) * T_ + (t + 1)) * H3 + (long)col * 4;
        async_b128_to_lds(dbase + (unsigned)c * 16, (const void*)g);
      }
    }

    // ---- gh = h @ Whh.T, all operands in LDS ----
    v8f acc[3] = {};
#pragma unroll
    for (int k = 0; k < 4; ++k) {
      v16h a = load_a16(h16, 128, lane, k * 32);
#pragma unroll
      for (int i = 0; i < 3; ++i) {
        v16h b = load_b16(Wl, 128, (w * 3 + i) * 16, k * 32, lane);
        acc[i] = WMMA_F16(a, b, acc[i]);
      }
    }
    int n = lane & 15, mo = (lane < 16) ? 0 : 8;
#pragma unroll
    for (int i = 0; i < 3; ++i) {
      int nb = (w * 3 + i) * 16;
#pragma unroll
      for (int r = 0; r < 8; ++r) gh[(mo + r) * H3 + nb + n] = acc[i][r];
    }

    // retire the (in-order) older async batch: gx(t) is now in LDS
    if (t + 1 < T_) wait_async6(); else wait_async0();
    __syncthreads();

    // ---- fused GRU gates (f32), gx read from LDS staging buffer ----
    const float* gxl = gxs + (t & 1) * 6144;    // 16*384 floats
#pragma unroll
    for (int jj = 0; jj < 8; ++jj) {
      int j = ecol + jj;
      float xr = gxl[erow * H3 + j];
      float xz = gxl[erow * H3 + 128 + j];
      float xn = gxl[erow * H3 + 256 + j];
      float hr = gh[erow * H3 + j]       + bhh[j];
      float hz = gh[erow * H3 + 128 + j] + bhh[128 + j];
      float hn = gh[erow * H3 + 256 + j] + bhh[256 + j];
      float rg = 1.0f / (1.0f + __expf(-(xr + hr)));
      float zg = 1.0f / (1.0f + __expf(-(xz + hz)));
      float ng = tanhf(xn + rg * hn);
      float hnew = (1.0f - zg) * ng + zg * h32[erow * H_ + j];
      h32[erow * H_ + j] = hnew;
      h16[erow * H_ + j] = (_Float16)hnew;
      hseq16[hrow + (long)t * H_ + j] = (_Float16)hnew;
      if (hseq32) hseq32[hrow + (long)t * H_ + j] = hnew;
    }
    __syncthreads();
  }
}

// ---------------------------------------------------------------------------
// Heads: s = h2 @ Ws.T + bs ; e = h2 @ We.T + be.
// ---------------------------------------------------------------------------
__global__ void k_heads(const _Float16* __restrict__ h2,
                        const _Float16* __restrict__ Ws16, const float* __restrict__ bs,
                        const _Float16* __restrict__ We16, const float* __restrict__ be,
                        float* __restrict__ souts, float* __restrict__ eouts) {
  int tid = threadIdx.x, lane = tid & 31, w = tid >> 5;
  long m0 = (long)blockIdx.x * 16;
  const _Float16* arow = h2 + m0 * 128;

  v8f acc[2] = {};
#pragma unroll
  for (int k = 0; k < 4; ++k) {
    v16h a  = load_a16(arow, 128, lane, k * 32);
    v16h b0 = load_b16(Ws16, 128, w * 16, k * 32, lane);
    acc[0] = WMMA_F16(a, b0, acc[0]);
    v16h b1 = load_b16(We16, 128, w * 16, k * 32, lane);
    acc[1] = WMMA_F16(a, b1, acc[1]);
  }
  int n = lane & 15, mo = (lane < 16) ? 0 : 8;
  int nb = w * 16;
  float bsv = bs[nb + n], bev = be[nb + n];
#pragma unroll
  for (int r = 0; r < 8; ++r) {
    souts[(m0 + mo + r) * 64 + nb + n] = acc[0][r] + bsv;
    eouts[(m0 + mo + r) * 64 + nb + n] = acc[1][r] + bev;
  }
}

// ---------------------------------------------------------------------------
extern "C" void kernel_launch(void* const* d_in, const int* in_sizes, int n_in,
                              void* d_out, int out_size, void* d_ws, size_t ws_size,
                              hipStream_t stream) {
  const float* x    = (const float*)d_in[0];
  const float* u    = (const float*)d_in[1];
  const float* W_in = (const float*)d_in[2];
  const float* b_in = (const float*)d_in[3];
  const float* Wih0 = (const float*)d_in[4];
  const float* Whh0 = (const float*)d_in[5];
  const float* bih0 = (const float*)d_in[6];
  const float* bhh0 = (const float*)d_in[7];
  const float* Wih1 = (const float*)d_in[8];
  const float* Whh1 = (const float*)d_in[9];
  const float* bih1 = (const float*)d_in[10];
  const float* bhh1 = (const float*)d_in[11];
  const float* Ws   = (const float*)d_in[12];
  const float* bs   = (const float*)d_in[13];
  const float* We   = (const float*)d_in[14];
  const float* be   = (const float*)d_in[15];

  char* ws = (char*)d_ws;
  float*     gx    = (float*)ws;                            // M*384 f32
  _Float16*  hbuf  = (_Float16*)(ws + (size_t)M_ * H3 * 4); // M*128 f16
  char* wp = ws + (size_t)M_ * H3 * 4 + (size_t)M_ * H_ * 2;
  _Float16* Win16   = (_Float16*)wp; wp += 128 * 96 * 2;
  _Float16* Wih0_16 = (_Float16*)wp; wp += H3 * H_ * 2;
  _Float16* Whh0_16 = (_Float16*)wp; wp += H3 * H_ * 2;
  _Float16* Wih1_16 = (_Float16*)wp; wp += H3 * H_ * 2;
  _Float16* Whh1_16 = (_Float16*)wp; wp += H3 * H_ * 2;
  _Float16* Ws16    = (_Float16*)wp; wp += 64 * H_ * 2;
  _Float16* We16    = (_Float16*)wp; wp += 64 * H_ * 2;

  // Allow 180KB dynamic LDS for the recurrent kernel (CDNA5: up to 320KB/WG).
  constexpr int GRU_LDS = 98304 + 24576 + 8192 + 4096 + 2 * 24576;
  (void)hipFuncSetAttribute((const void*)k_gru,
                            hipFuncAttributeMaxDynamicSharedMemorySize, GRU_LDS);

  // --- weight conversion (tiny) ---
  k_cvt_pad<<<(128 * 96 + 255) / 256, 256, 0, stream>>>(W_in, Win16, 128, 80, 96);
  k_cvt<<<(H3 * H_ + 255) / 256, 256, 0, stream>>>(Wih0, Wih0_16, H3 * H_);
  k_cvt<<<(H3 * H_ + 255) / 256, 256, 0, stream>>>(Whh0, Whh0_16, H3 * H_);
  k_cvt<<<(H3 * H_ + 255) / 256, 256, 0, stream>>>(Wih1, Wih1_16, H3 * H_);
  k_cvt<<<(H3 * H_ + 255) / 256, 256, 0, stream>>>(Whh1, Whh1_16, H3 * H_);
  k_cvt<<<(64 * H_ + 255) / 256, 256, 0, stream>>>(Ws, Ws16, 64 * H_);
  k_cvt<<<(64 * H_ + 255) / 256, 256, 0, stream>>>(We, We16, 64 * H_);

  float* outp  = (float*)d_out;
  float* s_out = outp;                       // [M,64]
  float* e_out = outp + (size_t)M_ * 64;     // [M,64]
  float* h_out = outp + (size_t)M_ * 128;    // [M,128]

  // --- pipeline ---
  k_xu<<<(int)(M_ / 16), 128, 0, stream>>>(x, u, Win16, b_in, hbuf);
  k_gx<<<(int)(M_ / 64), 256, 0, stream>>>(hbuf, Wih0_16, bih0, gx);
  k_gru<<<B_ / 16, 256, GRU_LDS, stream>>>(gx, Whh0_16, bhh0, hbuf, (float*)nullptr);
  k_gx<<<(int)(M_ / 64), 256, 0, stream>>>(hbuf, Wih1_16, bih1, gx);
  k_gru<<<B_ / 16, 256, GRU_LDS, stream>>>(gx, Whh1_16, bhh1, hbuf, h_out);
  k_heads<<<(int)(M_ / 16), 128, 0, stream>>>(hbuf, Ws16, bs, We16, be, s_out, e_out);
}
